// PairStr2Pair_10668698763285
// MI455X (gfx1250) — compile-verified, matrix operand order checked by reading, not run on hardware
//
#include <hip/hip_runtime.h>

// ============================================================================
// PairStr2Pair for MI455X (gfx1250): every matmul routed through
// v_wmma_f32_16x16x32_bf16 (wave32 WMMA, f32 accumulate).
//
// Pipeline (matches reference _forward):
//   1. state double-LN -> left/right (scalar, tiny)
//   2. rbf = sigmoid(outer(left,right) @ tgw + tgb) * (rbf_feat @ ew + eb)
//   3. four _subblock passes over 6 crops of 128x128 (tri-out, tri-in,
//      row-axial, col-axial), each gather -> op -> scatter-add/count -> avg
//   4. pair += FF(pair)
// Crop indices hardcoded from _crop_indices(256,128): nbox=4, pairs
// (0,1)(0,2)(0,3)(1,2)(1,3)(2,3), offsets all zero.
// ============================================================================

typedef __bf16 bf16;
typedef __attribute__((ext_vector_type(16))) bf16 v16bf;
typedef __attribute__((ext_vector_type(8)))  bf16 v8bf;
typedef __attribute__((ext_vector_type(8)))  float v8f;

#define LSEQ 256
#define DP   128
#define NC   6          // number of crops
#define CR   128        // crop size
#define NTOK (NC*CR*CR) // 98304 tokens per cropped tensor
#define PTOK (LSEQ*LSEQ)

// ---------------------------------------------------------------------------
// WMMA fragment loaders (16-bit A/B layout per CDNA5 ISA 7.12.2):
// lane r=lane&15 holds row r; halves 0..7 = K {ks..ks+7}, halves 8..15 =
// K {16+ks..23+ks}, ks = 8*(lane>>4).
// ---------------------------------------------------------------------------
__device__ __forceinline__ v16bf frag_rowmajor(const bf16* p, int ld) {
  int lane = threadIdx.x & 31;
  int r = lane & 15;
  int ks = (lane >> 4) << 3;
  v8bf lo = *(const v8bf*)(p + (size_t)r * ld + ks);
  v8bf hi = *(const v8bf*)(p + (size_t)r * ld + 16 + ks);
  v16bf f;
#pragma unroll
  for (int e = 0; e < 8; ++e) { f[e] = lo[e]; f[e + 8] = hi[e]; }
  return f;
}
// element (r,k) stored at p[k*ld + r]  (used for W[k][n] and transposed A)
__device__ __forceinline__ v16bf frag_colmajor(const bf16* p, int ld) {
  int lane = threadIdx.x & 31;
  int r = lane & 15;
  int ks = (lane >> 4) << 3;
  v16bf f;
#pragma unroll
  for (int e = 0; e < 8; ++e) {
    f[e]     = p[(size_t)(ks + e) * ld + r];
    f[e + 8] = p[(size_t)(16 + ks + e) * ld + r];
  }
  return f;
}

#define WMMA_BF16(a, b, c) \
  __builtin_amdgcn_wmma_f32_16x16x32_bf16(false, (a), false, (b), (short)0, (c), false, false)

// ---------------------------------------------------------------------------
// Generic GEMM: Y[M,N] = epi( (X[M,K] @ W[K,N]) * scale + bias[N] )
// X bf16 row-major, W f32 row-major (converted to bf16 while staging).
// EPI: 0 none, 1 sigmoid, 2 relu.  OUTBF: write bf16 instead of f32.
// Block = 256 threads (8 waves); 128(M) x <=128(N) per block; K chunked by 64.
// ---------------------------------------------------------------------------
template <int EPI, int OUTBF>
__global__ __launch_bounds__(256) void gemm_kernel(
    const bf16* __restrict__ X, const float* __restrict__ W,
    const float* __restrict__ bias, void* __restrict__ Y,
    int M, int N, int K, float scale) {
  __shared__ bf16 sX[128 * 64];   // 16 KB
  __shared__ bf16 sW[64 * 128];   // 16 KB
  int m0 = blockIdx.x * 128;
  int n0 = blockIdx.y * 128;
  int Nc = N - n0; if (Nc > 128) Nc = 128;
  int nT = (Nc + 15) >> 4;
  int tid = threadIdx.x, wave = tid >> 5, lane = tid & 31;
  v8f acc[8];
#pragma unroll
  for (int i = 0; i < 8; ++i) acc[i] = (v8f){0,0,0,0,0,0,0,0};

  for (int kc = 0; kc < K; kc += 64) {
    for (int idx = tid; idx < 128 * 64; idx += 256) {
      int r = idx >> 6, c = idx & 63;
      int mm = m0 + r, kk = kc + c;
      sX[idx] = (mm < M && kk < K) ? X[(size_t)mm * K + kk] : (bf16)0.0f;
    }
    for (int idx = tid; idx < 64 * 128; idx += 256) {
      int k = idx >> 7, c = idx & 127;
      int kk = kc + k, nn = n0 + c;
      sW[idx] = (kk < K && nn < N) ? (bf16)W[(size_t)kk * N + nn] : (bf16)0.0f;
    }
    __syncthreads();
#pragma unroll
    for (int kt = 0; kt < 2; ++kt) {
      v16bf a = frag_rowmajor(sX + wave * 16 * 64 + kt * 32, 64);
#pragma unroll
      for (int nt = 0; nt < 8; ++nt) {
        if (nt < nT) {
          v16bf b = frag_colmajor(sW + kt * 32 * 128 + nt * 16, 128);
          acc[nt] = WMMA_BF16(a, b, acc[nt]);
        }
      }
    }
    __syncthreads();
  }
  int nloc = lane & 15, mb = (lane >> 4) << 3;
#pragma unroll
  for (int nt = 0; nt < 8; ++nt) {
    if (nt >= nT) continue;
    int nn = n0 + nt * 16 + nloc;
    if (nn >= N) continue;
    float bv = bias ? bias[nn] : 0.0f;
#pragma unroll
    for (int r = 0; r < 8; ++r) {
      int mm = m0 + wave * 16 + mb + r;
      if (mm >= M) continue;
      float v = acc[nt][r] * scale + bv;
      if (EPI == 1) v = 1.0f / (1.0f + __expf(-v));
      if (EPI == 2) v = fmaxf(v, 0.0f);
      if (OUTBF) ((bf16*)Y)[(size_t)mm * N + nn] = (bf16)v;
      else       ((float*)Y)[(size_t)mm * N + nn] = v;
    }
  }
}

// ---------------------------------------------------------------------------
// Triangle einsum: per batch (crop n, channel d):
//   outgoing: O[i,j] = sum_k Lp[i,k] * Rp[j,k]
//   incoming: O[i,j] = sum_k Lp[k,i] * Rp[k,j]
// Lp/Rp layout: [(n*32+d)][128][128] bf16. Output O[((n*128+i)*128+j)*32+d].
// ---------------------------------------------------------------------------
template <int INC>
__global__ __launch_bounds__(256) void tri_gemm_kernel(
    const bf16* __restrict__ Lp, const bf16* __restrict__ Rp,
    float* __restrict__ O) {
  __shared__ bf16 sL[128 * 128];  // 32 KB
  __shared__ bf16 sR[128 * 128];  // 32 KB
  int batch = blockIdx.x;
  int nI = batch >> 5, d = batch & 31;
  const bf16* lb = Lp + (size_t)batch * 128 * 128;
  const bf16* rb = Rp + (size_t)batch * 128 * 128;
  int tid = threadIdx.x, wave = tid >> 5, lane = tid & 31;
  for (int idx = tid; idx < 128 * 128; idx += 256) { sL[idx] = lb[idx]; sR[idx] = rb[idx]; }
  __syncthreads();
  int jl = lane & 15, mb = (lane >> 4) << 3;
#pragma unroll 1
  for (int jt = 0; jt < 8; ++jt) {
    v8f acc = (v8f){0,0,0,0,0,0,0,0};
#pragma unroll
    for (int kt = 0; kt < 4; ++kt) {
      v16bf a = INC ? frag_colmajor(sL + kt * 32 * 128 + wave * 16, 128)
                    : frag_rowmajor(sL + wave * 16 * 128 + kt * 32, 128);
      v16bf b = INC ? frag_colmajor(sR + kt * 32 * 128 + jt * 16, 128)
                    : frag_rowmajor(sR + jt * 16 * 128 + kt * 32, 128);
      acc = WMMA_BF16(a, b, acc);
    }
#pragma unroll
    for (int r = 0; r < 8; ++r) {
      int i = wave * 16 + mb + r, j = jt * 16 + jl;
      O[((size_t)(nI * 128 + i) * 128 + j) * 32 + d] = acc[r];
    }
  }
}

// ---------------------------------------------------------------------------
// Attention logits: attn[b,i,j,h] = sum_n sum_k q[b,n,i,h,k]*k[b,n,j,h,k] + bh
// Q/K bf16 token-major [(b*128+n)*128+i][128]. grid (6,4).
// ---------------------------------------------------------------------------
__global__ __launch_bounds__(256) void attn_logits_kernel(
    const bf16* __restrict__ Q, const bf16* __restrict__ Km,
    const float* __restrict__ BH, float* __restrict__ LG) {
  __shared__ bf16 sQ[128 * 32];   // 8 KB
  __shared__ bf16 sK[128 * 32];   // 8 KB
  int b = blockIdx.x, h = blockIdx.y;
  int tid = threadIdx.x, wave = tid >> 5, lane = tid & 31;
  v8f acc[8];
#pragma unroll
  for (int i = 0; i < 8; ++i) acc[i] = (v8f){0,0,0,0,0,0,0,0};
  for (int n = 0; n < 128; ++n) {
    const bf16* qb = Q + ((size_t)(b * 128 + n) * 128) * 128 + h * 32;
    const bf16* kb = Km + ((size_t)(b * 128 + n) * 128) * 128 + h * 32;
    for (int idx = tid; idx < 128 * 32; idx += 256) {
      int r = idx >> 5, c = idx & 31;
      sQ[idx] = qb[(size_t)r * 128 + c];
      sK[idx] = kb[(size_t)r * 128 + c];
    }
    __syncthreads();
    v16bf a = frag_rowmajor(sQ + wave * 16 * 32, 32);
#pragma unroll
    for (int jt = 0; jt < 8; ++jt) {
      v16bf bb = frag_rowmajor(sK + jt * 16 * 32, 32);
      acc[jt] = WMMA_BF16(a, bb, acc[jt]);
    }
    __syncthreads();
  }
  int jl = lane & 15, mb = (lane >> 4) << 3;
#pragma unroll
  for (int jt = 0; jt < 8; ++jt)
#pragma unroll
    for (int r = 0; r < 8; ++r) {
      int i = wave * 16 + mb + r, j = jt * 16 + jl;
      LG[((size_t)(b * 4 + h) * 128 + i) * 128 + j] =
          acc[jt][r] + BH[((size_t)(b * 128 + i) * 128 + j) * 4 + h];
    }
}

// attn @ v: out[b,n,i,h*32+d] = sum_j attn[b,h,i,j] * v[b,n,j,h*32+d]
__global__ __launch_bounds__(256) void attn_v_kernel(
    const bf16* __restrict__ AT, const bf16* __restrict__ V,
    float* __restrict__ O) {
  __shared__ bf16 sV[128 * 128];  // 32 KB
  int b = blockIdx.x, n = blockIdx.y;
  const bf16* vb = V + ((size_t)(b * 128 + n) * 128) * 128;
  int tid = threadIdx.x, wave = tid >> 5, lane = tid & 31;
  for (int idx = tid; idx < 128 * 128; idx += 256) sV[idx] = vb[idx];
  __syncthreads();
  int nloc = lane & 15, mb = (lane >> 4) << 3;
  for (int h = 0; h < 4; ++h) {
    const bf16* ab = AT + ((size_t)(b * 4 + h) * 128 + wave * 16) * 128;
#pragma unroll 1
    for (int dt = 0; dt < 2; ++dt) {
      v8f acc = (v8f){0,0,0,0,0,0,0,0};
#pragma unroll
      for (int kt = 0; kt < 4; ++kt) {
        v16bf a = frag_rowmajor(ab + kt * 32, 128);
        v16bf bb = frag_colmajor(sV + kt * 32 * 128 + h * 32 + dt * 16, 128);
        acc = WMMA_BF16(a, bb, acc);
      }
#pragma unroll
      for (int r = 0; r < 8; ++r) {
        int i = wave * 16 + mb + r;
        int col = h * 32 + dt * 16 + nloc;
        O[((size_t)(b * 128 + n) * 128 + i) * 128 + col] = acc[r];
      }
    }
  }
}

// ---------------------------------------------------------------------------
// LayerNorm over C channels (one wave per token), bf16 output.
// ---------------------------------------------------------------------------
template <int C>
__global__ void ln_kernel(const float* __restrict__ X, const float* __restrict__ g,
                          const float* __restrict__ bta, bf16* __restrict__ Y,
                          int ntok) {
  int t = blockIdx.x * (blockDim.x >> 5) + (threadIdx.x >> 5);
  if (t >= ntok) return;
  int lane = threadIdx.x & 31;
  const int P = C / 32;
  float v[P > 0 ? P : 1];
  float s = 0.f;
#pragma unroll
  for (int i = 0; i < P; ++i) { v[i] = X[(size_t)t * C + i * 32 + lane]; s += v[i]; }
#pragma unroll
  for (int m = 16; m >= 1; m >>= 1) s += __shfl_xor(s, m);
  float mu = s / (float)C;
  float q = 0.f;
#pragma unroll
  for (int i = 0; i < P; ++i) { v[i] -= mu; q += v[i] * v[i]; }
#pragma unroll
  for (int m = 16; m >= 1; m >>= 1) q += __shfl_xor(q, m);
  float inv = rsqrtf(q / (float)C + 1e-5f);
#pragma unroll
  for (int i = 0; i < P; ++i)
    Y[(size_t)t * C + i * 32 + lane] =
        (bf16)(v[i] * inv * g[i * 32 + lane] + bta[i * 32 + lane]);
}

// Softmax over last dim (128), rows = 6*4*128, bf16 out.
__global__ void softmax_kernel(const float* __restrict__ LG, bf16* __restrict__ AT) {
  int row = blockIdx.x * (blockDim.x >> 5) + (threadIdx.x >> 5);
  if (row >= NC * 4 * 128) return;
  int lane = threadIdx.x & 31;
  const float* p = LG + (size_t)row * 128;
  float v[4], mx = -1e30f;
#pragma unroll
  for (int i = 0; i < 4; ++i) { v[i] = p[i * 32 + lane]; mx = fmaxf(mx, v[i]); }
#pragma unroll
  for (int m = 16; m >= 1; m >>= 1) mx = fmaxf(mx, __shfl_xor(mx, m));
  float s = 0.f;
#pragma unroll
  for (int i = 0; i < 4; ++i) { v[i] = __expf(v[i] - mx); s += v[i]; }
#pragma unroll
  for (int m = 16; m >= 1; m >>= 1) s += __shfl_xor(s, m);
  float inv = 1.0f / s;
#pragma unroll
  for (int i = 0; i < 4; ++i)
    AT[(size_t)row * 128 + i * 32 + lane] = (bf16)(v[i] * inv);
}

// ---------------------------------------------------------------------------
// State path: st = LN(LN(state)); left = st@plw+plb; right = st@prw+prb.
// One wave per token (32 channels).
// ---------------------------------------------------------------------------
__global__ void state_kernel(const float* __restrict__ st_in,
                             const float* __restrict__ nsg, const float* __restrict__ nsb,
                             const float* __restrict__ plw, const float* __restrict__ plb,
                             const float* __restrict__ prw, const float* __restrict__ prb,
                             float* __restrict__ left, float* __restrict__ right) {
  int t = blockIdx.x * (blockDim.x >> 5) + (threadIdx.x >> 5);
  if (t >= LSEQ) return;
  int c = threadIdx.x & 31;
  float v = st_in[t * 32 + c];
#pragma unroll
  for (int rep = 0; rep < 2; ++rep) {
    float s = v;
#pragma unroll
    for (int m = 16; m >= 1; m >>= 1) s += __shfl_xor(s, m);
    float mu = s * (1.0f / 32.0f);
    float d = v - mu;
    float q = d * d;
#pragma unroll
    for (int m = 16; m >= 1; m >>= 1) q += __shfl_xor(q, m);
    v = d * rsqrtf(q * (1.0f / 32.0f) + 1e-5f) * nsg[c] + nsb[c];
  }
  int o = c & 15;
  const float* W = (c < 16) ? plw : prw;
  float acc = (c < 16) ? plb[o] : prb[o];
  for (int k = 0; k < 32; ++k) acc += __shfl(v, k) * W[k * 16 + o];
  if (c < 16) left[t * 16 + o] = acc; else right[t * 16 + o] = acc;
}

// ---------------------------------------------------------------------------
// Elementwise / gather / scatter kernels
// ---------------------------------------------------------------------------
__device__ __forceinline__ int gmap(int n, int c) {
  // crop pairs: 0:(0,1) 1:(0,2) 2:(0,3) 3:(1,2) 4:(1,3) 5:(2,3)
  int i0, i1;
  if (n < 3)      { i0 = 0; i1 = n + 1; }
  else if (n < 5) { i0 = 1; i1 = n - 1; }
  else            { i0 = 2; i1 = 3; }
  int blk = (c < 64) ? i0 : i1;
  return (blk << 6) | (c & 63);
}

__global__ void gather_kernel(const float4* __restrict__ SRC, float4* __restrict__ DST,
                              int trans) {
  int idx = blockIdx.x * blockDim.x + threadIdx.x;  // NTOK*32 float4
  if (idx >= NTOK * 32) return;
  int j4 = idx & 31;
  int t = idx >> 5;
  int bcol = t & 127, arow = (t >> 7) & 127, n = t >> 14;
  int gi = gmap(n, trans ? bcol : arow);
  int gj = gmap(n, trans ? arow : bcol);
  DST[idx] = SRC[((size_t)(gi * LSEQ + gj)) * 32 + j4];
}

__global__ void scatter_kernel(const float* __restrict__ SRC, float* __restrict__ ACC,
                               int trans) {
  int idx = blockIdx.x * blockDim.x + threadIdx.x;  // NTOK*128
  if (idx >= NTOK * 128) return;
  int d = idx & 127;
  int t = idx >> 7;
  int bcol = t & 127, arow = (t >> 7) & 127, n = t >> 14;
  int gi = gmap(n, trans ? bcol : arow);
  int gj = gmap(n, trans ? arow : bcol);
  atomicAdd(&ACC[((size_t)(gi * LSEQ + gj)) * DP + d], SRC[idx]);
}

__global__ void cnt_kernel(float* __restrict__ CNT) {
  int t = blockIdx.x * blockDim.x + threadIdx.x;  // NC*128*128
  if (t >= NC * CR * CR) return;
  int bcol = t & 127, arow = (t >> 7) & 127, n = t >> 14;
  atomicAdd(&CNT[gmap(n, arow) * LSEQ + gmap(n, bcol)], 1.0f);
}

__global__ void combine_kernel(float4* __restrict__ PW, const float4* __restrict__ ACC,
                               const float* __restrict__ CNT) {
  int idx = blockIdx.x * blockDim.x + threadIdx.x;  // PTOK*32
  if (idx >= PTOK * 32) return;
  float inv = 1.0f / CNT[idx >> 5];
  float4 a = ACC[idx];
  float4 p = PW[idx];
  p.x += a.x * inv; p.y += a.y * inv; p.z += a.z * inv; p.w += a.w * inv;
  PW[idx] = p;
}

__global__ void zero4_kernel(float4* __restrict__ P, int n4) {
  int i = blockIdx.x * blockDim.x + threadIdx.x;
  if (i < n4) P[i] = (float4){0.f, 0.f, 0.f, 0.f};
}
__global__ void copy4_kernel(const float4* __restrict__ S, float4* __restrict__ D, int n4) {
  int i = blockIdx.x * blockDim.x + threadIdx.x;
  if (i < n4) D[i] = S[i];
}
__global__ void add4_kernel(const float4* __restrict__ A, const float4* __restrict__ B,
                            float4* __restrict__ D, int n4) {
  int i = blockIdx.x * blockDim.x + threadIdx.x;
  if (i >= n4) return;
  float4 a = A[i], b = B[i];
  D[i] = (float4){a.x + b.x, a.y + b.y, a.z + b.z, a.w + b.w};
}
__global__ void mul4_kernel(const float4* __restrict__ A, const float4* __restrict__ B,
                            float4* __restrict__ D, int n4) {
  int i = blockIdx.x * blockDim.x + threadIdx.x;
  if (i >= n4) return;
  float4 a = A[i], b = B[i];
  D[i] = (float4){a.x * b.x, a.y * b.y, a.z * b.z, a.w * b.w};
}
__global__ void mulbf_kernel(const float* __restrict__ A, const float* __restrict__ B,
                             bf16* __restrict__ D, int n) {
  int i = blockIdx.x * blockDim.x + threadIdx.x;
  if (i < n) D[i] = (bf16)(A[i] * B[i]);
}
__global__ void packbf_kernel(const float* __restrict__ A, bf16* __restrict__ D, int n) {
  int i = blockIdx.x * blockDim.x + threadIdx.x;
  if (i < n) D[i] = (bf16)A[i];
}
// left/right product -> channel-major bf16 for tri GEMM:
// OUT[(n*32+d)][i][k] = SG[t*32+d]*P[t*32+d]*scale,  t = (n*128+i)*128+k
__global__ void tri_pack_kernel(const float* __restrict__ SG, const float* __restrict__ P,
                                bf16* __restrict__ OUT, float scale) {
  int idx = blockIdx.x * blockDim.x + threadIdx.x;  // NTOK*32
  if (idx >= NTOK * 32) return;
  int d = idx & 31;
  int t = idx >> 5;
  int k = t & 127, i = (t >> 7) & 127, n = t >> 14;
  OUT[(((size_t)(n * 32 + d) * 128 + i) << 7) + k] = (bf16)(SG[idx] * P[idx] * scale);
}
// gate outer product: OUT[t*256 + a*16 + b] = left[i,a]*right[j,b],  t=i*256+j
__global__ void gate_outer_kernel(const float* __restrict__ left,
                                  const float* __restrict__ right,
                                  bf16* __restrict__ OUT) {
  int idx = blockIdx.x * blockDim.x + threadIdx.x;  // PTOK*256
  if (idx >= PTOK * 256) return;
  int c = idx & 255;
  int t = idx >> 8;
  int i = t >> 8, j = t & 255;
  OUT[idx] = (bf16)(left[i * 16 + (c >> 4)] * right[j * 16 + (c & 15)]);
}

// ============================================================================
// Host side
// ============================================================================
struct TmP { const float *gb,*gw,*lgb,*lgw,*lpb,*lpw,*nb,*ng,*nob,*nog,*ob,*ow,*rgb,*rgw,*rpb,*rpw; };
struct AxP { const float *bw,*gb,*gw,*kw,*nbb,*nbg,*npb,*npg,*ob,*ow,*qw,*vw; };

extern "C" void kernel_launch(void* const* d_in, const int* in_sizes, int n_in,
                              void* d_out, int out_size, void* d_ws, size_t ws_size,
                              hipStream_t stream) {
  (void)in_sizes; (void)n_in; (void)out_size; (void)ws_size;
  // Top-level inputs in setup_inputs() insertion order; params flattened in
  // jax tree (sorted-key, recursive) order:
  //  +0..11  col.{bw,gb,gw,kw,nbb,nbg,npb,npg,ob,ow,qw,vw}
  //  +12 eb  +13 ew
  //  +14..19 ff.{b1,b2,nb,ng,w1,w2}
  //  +20 nsb +21 nsg +22 plb +23 plw +24 prb +25 prw
  //  +26..37 row.{bw,gb,gw,kw,nbb,nbg,npb,npg,ob,ow,qw,vw}
  //  +38 tgb +39 tgw
  //  +40..55 tmi.{gb,gw,lgb,lgw,lpb,lpw,nb,ng,nob,nog,ob,ow,rgb,rgw,rpb,rpw}
  //  +56..71 tmo.{...same...}
  const float* pair     = (const float*)d_in[0];
  const float* rbf_feat = (const float*)d_in[1];
  const float* state    = (const float*)d_in[2];
  auto P = [&](int i) { return (const float*)d_in[3 + i]; };
  AxP col = {P(0),P(1),P(2),P(3),P(4),P(5),P(6),P(7),P(8),P(9),P(10),P(11)};
  const float *eb = P(12), *ew = P(13);
  const float *ff_b1 = P(14), *ff_b2 = P(15), *ff_nb = P(16), *ff_ng = P(17),
              *ff_w1 = P(18), *ff_w2 = P(19);
  const float *nsb = P(20), *nsg = P(21), *plb = P(22), *plw = P(23),
              *prb = P(24), *prw = P(25);
  AxP row = {P(26),P(27),P(28),P(29),P(30),P(31),P(32),P(33),P(34),P(35),P(36),P(37)};
  const float *tgb = P(38), *tgw = P(39);
  TmP tmi = {P(40),P(41),P(42),P(43),P(44),P(45),P(46),P(47),P(48),P(49),P(50),P(51),P(52),P(53),P(54),P(55)};
  TmP tmo = {P(56),P(57),P(58),P(59),P(60),P(61),P(62),P(63),P(64),P(65),P(66),P(67),P(68),P(69),P(70),P(71)};

  // ------- workspace layout -------
  char* w = (char*)d_ws;
  auto alloc = [&](size_t bytes) { char* p = w; w += (bytes + 255) & ~(size_t)255; return p; };
  float* PAIRW = (float*)alloc((size_t)PTOK * DP * 4);
  float* RBF   = (float*)alloc((size_t)PTOK * DP * 4);
  float* ACC   = (float*)alloc((size_t)PTOK * DP * 4);
  float* CNT   = (float*)alloc((size_t)PTOK * 4);
  float* PC    = (float*)alloc((size_t)NTOK * DP * 4);
  float* RC    = (float*)alloc((size_t)NTOK * DP * 4);
  float* OPOUT = (float*)alloc((size_t)NTOK * DP * 4);   // also reused as bf16 V
  bf16*  XLN   = (bf16*) alloc((size_t)NTOK * DP * 2);
  float* S1    = (float*)alloc((size_t)NTOK * DP * 4);
  float* S2    = (float*)alloc((size_t)NTOK * DP * 4);
  bf16*  T1    = (bf16*) alloc((size_t)PTOK * 256 * 2);
  bf16*  T2    = (bf16*) alloc((size_t)PTOK * 256 * 2);
  bf16*  T3    = (bf16*) alloc((size_t)PTOK * 256 * 2);
  float* LEFT  = (float*)alloc((size_t)LSEQ * 16 * 4);
  float* RIGHT = (float*)alloc((size_t)LSEQ * 16 * 4);
  float* BH    = (float*)alloc((size_t)NTOK * 4 * 4);
  float* LG    = (float*)alloc((size_t)NC * 4 * 128 * 128 * 4);
  bf16*  ATW   = (bf16*) alloc((size_t)NC * 4 * 128 * 128 * 2);

  auto blks = [](long n, int bs) { return (unsigned)((n + bs - 1) / bs); };
  auto gemm = [&](int epi, int outbf, const bf16* X, const float* W_, const float* B_,
                  void* Y, int M, int N, int K, float scale) {
    dim3 g(blks(M, 128), blks(N, 128));
    if      (epi == 0 && !outbf) gemm_kernel<0,0><<<g,256,0,stream>>>(X,W_,B_,Y,M,N,K,scale);
    else if (epi == 1 && !outbf) gemm_kernel<1,0><<<g,256,0,stream>>>(X,W_,B_,Y,M,N,K,scale);
    else if (epi == 0 &&  outbf) gemm_kernel<0,1><<<g,256,0,stream>>>(X,W_,B_,Y,M,N,K,scale);
    else                         gemm_kernel<2,1><<<g,256,0,stream>>>(X,W_,B_,Y,M,N,K,scale);
  };

  const long PAIR4 = (long)PTOK * 32;   // float4 count of pair tensor
  const long CROP4 = (long)NTOK * 32;

  // ------- setup: pair copy, state path, gated rbf, counts -------
  copy4_kernel<<<blks(PAIR4,256),256,0,stream>>>((const float4*)pair, (float4*)PAIRW, PAIR4);
  state_kernel<<<blks(LSEQ,8),256,0,stream>>>(state, nsg, nsb, plw, plb, prw, prb, LEFT, RIGHT);
  packbf_kernel<<<blks((long)PTOK*64,256),256,0,stream>>>(rbf_feat, T1, PTOK*64);
  gemm(0,0, T1, ew, eb, S1, PTOK, DP, 64, 1.0f);                       // rbf linear
  gate_outer_kernel<<<blks((long)PTOK*256,256),256,0,stream>>>(LEFT, RIGHT, T2);
  gemm(1,0, T2, tgw, tgb, S2, PTOK, DP, 256, 1.0f);                    // sigmoid gate
  mul4_kernel<<<blks(PAIR4,256),256,0,stream>>>((const float4*)S2,(const float4*)S1,(float4*)RBF,PAIR4);
  zero4_kernel<<<blks(PTOK/4,256),256,0,stream>>>((float4*)CNT, PTOK/4);
  cnt_kernel<<<blks(NC*CR*CR,256),256,0,stream>>>(CNT);

  // ------- triangle multiplication subblocks -------
  auto tri_subblock = [&](const TmP& pp, bool incoming) {
    zero4_kernel<<<blks(PAIR4,256),256,0,stream>>>((float4*)ACC, PAIR4);
    gather_kernel<<<blks(CROP4,256),256,0,stream>>>((const float4*)PAIRW,(float4*)PC,0);
    ln_kernel<128><<<blks(NTOK,8),256,0,stream>>>(PC, pp.ng, pp.nb, XLN, NTOK);
    gemm(1,0, XLN, pp.lgw, pp.lgb, S1, NTOK, 32, 128, 1.0f);
    gemm(0,0, XLN, pp.lpw, pp.lpb, S2, NTOK, 32, 128, 1.0f);
    tri_pack_kernel<<<blks((long)NTOK*32,256),256,0,stream>>>(S1, S2, T1, 1.0f);
    gemm(1,0, XLN, pp.rgw, pp.rgb, S1, NTOK, 32, 128, 1.0f);
    gemm(0,0, XLN, pp.rpw, pp.rpb, S2, NTOK, 32, 128, 1.0f);
    tri_pack_kernel<<<blks((long)NTOK*32,256),256,0,stream>>>(S1, S2, T2, 1.0f/(float)CR);
    if (incoming) tri_gemm_kernel<1><<<NC*32,256,0,stream>>>(T1, T2, S1);
    else          tri_gemm_kernel<0><<<NC*32,256,0,stream>>>(T1, T2, S1);
    ln_kernel<32><<<blks(NTOK,8),256,0,stream>>>(S1, pp.nog, pp.nob, T3, NTOK);
    gemm(0,0, T3, pp.ow, pp.ob, S2, NTOK, 128, 32, 1.0f);
    gemm(1,0, XLN, pp.gw, pp.gb, S1, NTOK, 128, 128, 1.0f);
    mul4_kernel<<<blks(CROP4,256),256,0,stream>>>((const float4*)S1,(const float4*)S2,(float4*)OPOUT,CROP4);
    scatter_kernel<<<blks((long)NTOK*128,256),256,0,stream>>>(OPOUT, ACC, 0);
    combine_kernel<<<blks(PAIR4,256),256,0,stream>>>((float4*)PAIRW,(const float4*)ACC,CNT);
  };
  tri_subblock(tmo, false);   // outgoing
  tri_subblock(tmi, true);    // incoming

  // ------- axial attention subblocks -------
  auto ax_subblock = [&](const AxP& pp, int trans) {
    zero4_kernel<<<blks(PAIR4,256),256,0,stream>>>((float4*)ACC, PAIR4);
    gather_kernel<<<blks(CROP4,256),256,0,stream>>>((const float4*)PAIRW,(float4*)PC,trans);
    gather_kernel<<<blks(CROP4,256),256,0,stream>>>((const float4*)RBF,(float4*)RC,trans);
    ln_kernel<128><<<blks(NTOK,8),256,0,stream>>>(PC, pp.npg, pp.npb, XLN, NTOK);
    ln_kernel<128><<<blks(NTOK,8),256,0,stream>>>(RC, pp.nbg, pp.nbb, T3, NTOK);
    bf16* Vbf = (bf16*)OPOUT;
    gemm(0,1, XLN, pp.qw, nullptr, T1,  NTOK, 128, 128, 0.17677669529f); // 1/sqrt(32)
    gemm(0,1, XLN, pp.kw, nullptr, T2,  NTOK, 128, 128, 1.0f/(float)CR);
    gemm(0,1, XLN, pp.vw, nullptr, Vbf, NTOK, 128, 128, 1.0f);
    gemm(1,0, XLN, pp.gw, pp.gb,   S1,  NTOK, 128, 128, 1.0f);
    gemm(0,0, T3,  pp.bw, nullptr, BH,  NTOK, 4,   128, 1.0f);
    attn_logits_kernel<<<dim3(NC,4),256,0,stream>>>(T1, T2, BH, LG);
    softmax_kernel<<<blks(NC*4*128,8),256,0,stream>>>(LG, ATW);
    attn_v_kernel<<<dim3(NC,128),256,0,stream>>>(ATW, Vbf, S2);
    mulbf_kernel<<<blks((long)NTOK*128,256),256,0,stream>>>(S1, S2, T3, NTOK*128);
    gemm(0,0, T3, pp.ow, pp.ob, OPOUT, NTOK, 128, 128, 1.0f);
    scatter_kernel<<<blks((long)NTOK*128,256),256,0,stream>>>(OPOUT, ACC, trans);
    combine_kernel<<<blks(PAIR4,256),256,0,stream>>>((float4*)PAIRW,(const float4*)ACC,CNT);
  };
  ax_subblock(row, 1);  // row: operate in transposed crop space
  ax_subblock(col, 0);  // col

  // ------- feed-forward + final output -------
  ln_kernel<128><<<blks(PTOK,8),256,0,stream>>>(PAIRW, ff_ng, ff_nb, XLN, PTOK);
  gemm(2,1, XLN, ff_w1, ff_b1, T1, PTOK, 256, 128, 1.0f);  // relu, bf16 out
  gemm(0,0, T1,  ff_w2, ff_b2, S1, PTOK, 128, 256, 1.0f);
  add4_kernel<<<blks(PAIR4,256),256,0,stream>>>((const float4*)PAIRW,(const float4*)S1,
                                                (float4*)d_out, PAIR4);
}
